// STAR_43147241455861
// MI455X (gfx1250) — matrix operand order repaired and checked by reading, not compiled, not run on hardware
//
#include <hip/hip_runtime.h>
#include <hip/hip_bf16.h>

// ---------------------------------------------------------------------------
// Problem constants
// ---------------------------------------------------------------------------
#define TT     8192
#define LL     32
#define HH     128
#define OBSN   128
#define INN    160
#define HIDN   1024
#define FEATN  512

typedef __attribute__((ext_vector_type(16))) _Float16 v16h;
typedef __attribute__((ext_vector_type(8)))  _Float16 v8h;
typedef __attribute__((ext_vector_type(8)))  float    v8f;
typedef __attribute__((ext_vector_type(4)))  int      v4i;

typedef __attribute__((address_space(1))) v4i as1_v4i;
typedef __attribute__((address_space(3))) v4i as3_v4i;

__device__ __forceinline__ float sigmf(float x) { return 1.0f / (1.0f + __expf(-x)); }

// ---------------------------------------------------------------------------
// CDNA5 async global->LDS copy (16 bytes per lane) + ASYNCcnt wait.
// Builtin signature (from compiler diagnostic): (v4i* AS1, v4i* AS3, Ii, Ii).
// ---------------------------------------------------------------------------
__device__ __forceinline__ void cp16_async(const _Float16* g, _Float16* l) {
#if defined(__gfx1250__) && __has_builtin(__builtin_amdgcn_global_load_async_to_lds_b128)
    __builtin_amdgcn_global_load_async_to_lds_b128(
        (as1_v4i*)(g), (as3_v4i*)(l), 0, 0);
#else
    *reinterpret_cast<float4*>(l) = *reinterpret_cast<const float4*>(g);
#endif
}

template <int N>
__device__ __forceinline__ void async_wait_le() {
#if defined(__gfx1250__) && __has_builtin(__builtin_amdgcn_s_wait_asynccnt)
    __builtin_amdgcn_s_wait_asynccnt(N);
#elif defined(__gfx1250__)
    asm volatile("s_wait_asynccnt %0" :: "i"(N) : "memory");
#endif
}

// ---------------------------------------------------------------------------
// Flags: is_b[t] = (obs[t,2] == 1.0f)
// ---------------------------------------------------------------------------
__global__ void k_flags(const float* __restrict__ obs, int* __restrict__ flags) {
    int t = blockIdx.x * blockDim.x + threadIdx.x;
    if (t < TT) flags[t] = (obs[(size_t)t * OBSN + 2] == 1.0f) ? 1 : 0;
}

// ---------------------------------------------------------------------------
// Gate-input precompute (parallel over T):
//   z = xl @ W1^T + b1            (128)
//   Gx[t] = z @ Wih^T + bih + bhh (512)     -- the h-independent gate input
// ---------------------------------------------------------------------------
__global__ __launch_bounds__(512) void k_gate_pre(
    const float* __restrict__ obs,
    const float* __restrict__ W1,  const float* __restrict__ b1,
    const float* __restrict__ Wih, const float* __restrict__ bih,
    const float* __restrict__ bhh,
    float* __restrict__ Gx)
{
    __shared__ float W1T[32 * 128];   // W1T[k*128 + j] = W1[j][k]
    __shared__ float xbuf[32];
    __shared__ float zbuf[128];
    const int tid = threadIdx.x;

    for (int idx = tid; idx < 32 * 128; idx += 512) {
        int j = idx >> 5, k = idx & 31;
        W1T[k * 128 + j] = W1[idx];
    }
    float4 w4[32];
    {
        const float4* wr = reinterpret_cast<const float4*>(Wih + (size_t)tid * HH);
        #pragma unroll
        for (int i = 0; i < 32; ++i) w4[i] = wr[i];
    }
    const float bb = bih[tid] + bhh[tid];
    __syncthreads();

    const int t0 = blockIdx.x * 64;
    for (int tt = 0; tt < 64; ++tt) {
        const int t = t0 + tt;
        if (tid < 32) xbuf[tid] = obs[(size_t)t * OBSN + tid];
        __syncthreads();
        if (tid < 128) {
            float z = b1[tid];
            #pragma unroll
            for (int k = 0; k < 32; ++k) z += W1T[k * 128 + tid] * xbuf[k];
            zbuf[tid] = z;
        }
        __syncthreads();
        {
            const float4* z4 = reinterpret_cast<const float4*>(zbuf);
            float s = bb;
            #pragma unroll
            for (int i = 0; i < 32; ++i) {
                float4 h = z4[i];
                s += w4[i].x * h.x + w4[i].y * h.y + w4[i].z * h.z + w4[i].w * h.w;
            }
            Gx[(size_t)t * 512 + tid] = s;
        }
        __syncthreads();
    }
}

// ---------------------------------------------------------------------------
// Sequential LSTM scan: 3 independent chains, one persistent WG each.
// mode 0: always commit; mode 1: commit if flag; mode 2: commit if !flag.
// Whh row (128 f32) lives in VGPRs; h broadcast from LDS.
// ---------------------------------------------------------------------------
__global__ __launch_bounds__(512) void k_lstm_scan(
    const float* __restrict__ Whh0, const float* __restrict__ Whh1, const float* __restrict__ Whh2,
    const float* __restrict__ W2_0, const float* __restrict__ W2_1, const float* __restrict__ W2_2,
    const float* __restrict__ b2_0, const float* __restrict__ b2_1, const float* __restrict__ b2_2,
    const float* __restrict__ Gx,   // [3][T][512]
    float* __restrict__ Ovec,       // [3][T][32]
    const int* __restrict__ flags)
{
    __shared__ float W2T[128 * 32];  // W2T[k*32 + j] = W2[j][k]
    __shared__ float g[512];
    __shared__ float hbuf[128], cbuf[128], hnew[128];
    const int tid  = threadIdx.x;
    const int mode = blockIdx.x;

    const float* Whh = (mode == 0) ? Whh0 : (mode == 1) ? Whh1 : Whh2;
    const float* W2  = (mode == 0) ? W2_0 : (mode == 1) ? W2_1 : W2_2;
    const float* b2  = (mode == 0) ? b2_0 : (mode == 1) ? b2_1 : b2_2;
    const float* GxC = Gx + (size_t)mode * TT * 512;
    float*       Oc  = Ovec + (size_t)mode * TT * 32;

    for (int idx = tid; idx < 32 * 128; idx += 512) {
        int j = idx >> 7, k = idx & 127;
        W2T[k * 32 + j] = W2[idx];
    }
    if (tid < 128) { hbuf[tid] = 0.0f; cbuf[tid] = 0.0f; }
    float4 w4[32];
    {
        const float4* wr = reinterpret_cast<const float4*>(Whh + (size_t)tid * HH);
        #pragma unroll
        for (int i = 0; i < 32; ++i) w4[i] = wr[i];
    }
    const float bias2 = (tid < 32) ? b2[tid] : 0.0f;
    __syncthreads();

    for (int t = 0; t < TT; ++t) {
        {   // g = Gx[t] + Whh @ h      (512 dots of length 128)
            const float4* h4 = reinterpret_cast<const float4*>(hbuf);
            float s = GxC[(size_t)t * 512 + tid];
            #pragma unroll
            for (int i = 0; i < 32; ++i) {
                float4 h = h4[i];
                s += w4[i].x * h.x + w4[i].y * h.y + w4[i].z * h.z + w4[i].w * h.w;
            }
            g[tid] = s;
        }
        __syncthreads();
        if (tid < 128) {
            float gi = g[tid], gf = g[tid + 128], gg = g[tid + 256], go = g[tid + 384];
            float cn = sigmf(gf) * cbuf[tid] + sigmf(gi) * tanhf(gg);
            float hn = sigmf(go) * tanhf(cn);
            hnew[tid] = hn;
            int  fl = flags[t];
            bool commit = (mode == 0) || (mode == 1 && fl) || (mode == 2 && !fl);
            if (commit) { hbuf[tid] = hn; cbuf[tid] = cn; }
        }
        __syncthreads();
        if (tid < 32) {  // o = hnew @ W2^T + b2
            float s = bias2;
            #pragma unroll 4
            for (int k = 0; k < 128; ++k) s += W2T[k * 32 + tid] * hnew[k];
            Oc[(size_t)t * 32 + tid] = s;
        }
        // next-iteration first barrier protects hnew/g reuse
    }
}

// ---------------------------------------------------------------------------
// "Attention": att = softmax( (o @ Wv^T + bv) @ Wo^T + bo )  row-wise (32)
// ---------------------------------------------------------------------------
__global__ __launch_bounds__(256) void k_attention(
    const float* __restrict__ O,  const float* __restrict__ Wv, const float* __restrict__ bv,
    const float* __restrict__ Wo, const float* __restrict__ bo, float* __restrict__ att)
{
    __shared__ float WoT[256 * 32];  // WoT[m*32 + j] = Wo[j][m]
    __shared__ float vbuf[256];
    __shared__ float xb[32];
    const int tid = threadIdx.x;

    for (int idx = tid; idx < 32 * 256; idx += 256) {
        int j = idx >> 8, m = idx & 255;
        WoT[m * 32 + j] = Wo[idx];
    }
    float wv[32];
    {
        const float* p = Wv + (size_t)tid * 32;
        #pragma unroll
        for (int l = 0; l < 32; ++l) wv[l] = p[l];
    }
    const float bvm = bv[tid];
    __syncthreads();

    const int t0 = blockIdx.x * 32;
    for (int tt = 0; tt < 32; ++tt) {
        const int t = t0 + tt;
        if (tid < 32) xb[tid] = O[(size_t)t * 32 + tid];
        __syncthreads();
        float v = bvm;
        #pragma unroll
        for (int l = 0; l < 32; ++l) v += wv[l] * xb[l];
        vbuf[tid] = v;
        __syncthreads();
        if (tid < 32) {
            float a = bo[tid];
            for (int m = 0; m < 256; ++m) a += WoT[m * 32 + tid] * vbuf[m];
            float mx = a;
            #pragma unroll
            for (int off = 16; off > 0; off >>= 1) mx = fmaxf(mx, __shfl_xor(mx, off, 32));
            float e  = __expf(a - mx);
            float sm = e;
            #pragma unroll
            for (int off = 16; off > 0; off >>= 1) sm += __shfl_xor(sm, off, 32);
            att[(size_t)t * 32 + tid] = e / sm;
        }
        __syncthreads();
    }
}

// ---------------------------------------------------------------------------
// f32 -> f16 conversion (weights)
// ---------------------------------------------------------------------------
__global__ void k_f32tof16(const float* __restrict__ src, _Float16* __restrict__ dst, int n) {
    int i = blockIdx.x * blockDim.x + threadIdx.x;
    if (i < n) dst[i] = (_Float16)src[i];
}

// ---------------------------------------------------------------------------
// X = concat(att, obs) as f16  [T,160]
// ---------------------------------------------------------------------------
__global__ void k_build_x(const float* __restrict__ att, const float* __restrict__ obs,
                          _Float16* __restrict__ X) {
    int i = blockIdx.x * blockDim.x + threadIdx.x;
    if (i < TT * INN) {
        int t = i / INN, c = i - t * INN;
        float v = (c < LL) ? att[(size_t)t * LL + c] : obs[(size_t)t * OBSN + (c - LL)];
        X[i] = (_Float16)v;
    }
}

// ---------------------------------------------------------------------------
// WMMA f16 GEMM:  C[M,N] = A[M,K] @ W[N,K]^T  (+bias, opt ReLU)
// Block = 8 waves: 128(M) x 128(N) tile.  K-step 32.
// A/B tiles staged in LDS (padded stride) via async global->LDS copies with
// double buffering: tile k+1 streams in (ASYNCcnt) while WMMAs consume tile k.
// ---------------------------------------------------------------------------
#define SROW 40   // padded LDS row stride in halves (32 data + 8 pad)

__global__ __launch_bounds__(256) void k_gemm_wmma(
    const _Float16* __restrict__ A, const _Float16* __restrict__ W,
    const float* __restrict__ bias, int M, int N, int K, int relu,
    _Float16* __restrict__ out16, float* __restrict__ out32,
    int ld_out, int col_off)
{
    __shared__ _Float16 Asm[2][128 * SROW];
    __shared__ _Float16 Bsm[2][128 * SROW];

    const int tid  = threadIdx.x;
    const int lane = tid & 31;
    const int wave = tid >> 5;
    const int m0   = blockIdx.y * 128 + wave * 16;
    const int n0   = blockIdx.x * 128;
    const int half = lane >> 4;
    const int l15  = lane & 15;

    const _Float16* Ablk = A + (size_t)(blockIdx.y * 128) * K;
    const _Float16* Wblk = W + (size_t)n0 * K;

    // stage one 128x32 A tile + 128x32 B tile into LDS buffer b (k-offset k0)
    auto stage = [&](int k0, int b) {
        _Float16* lA = &Asm[b][0];
        _Float16* lB = &Bsm[b][0];
        #pragma unroll
        for (int q = 0; q < 2; ++q) {
            int idx = (tid << 1) | q;        // 0..511 segments (row, 8-half seg)
            int r = idx >> 2, s = idx & 3;
            cp16_async(Ablk + (size_t)r * K + k0 + s * 8, lA + r * SROW + s * 8);
            cp16_async(Wblk + (size_t)r * K + k0 + s * 8, lB + r * SROW + s * 8);
        }
    };

    v8f acc[8] = {};
    const int nk = K >> 5;

    stage(0, 0);
    int buf = 0;
    for (int ks = 0; ks < nk; ++ks) {
        if (ks + 1 < nk) {
            stage((ks + 1) << 5, buf ^ 1);
            async_wait_le<4>();   // group k resident; k+1 (4 ops/wave) in flight
        } else {
            async_wait_le<0>();
        }
        __syncthreads();

        // A fragment from LDS
        const int rbase = (wave * 16 + l15) * SROW;
        v8h alo = *reinterpret_cast<const v8h*>(&Asm[buf][rbase + (half ? 8 : 0)]);
        v8h ahi = *reinterpret_cast<const v8h*>(&Asm[buf][rbase + (half ? 24 : 16)]);
        v16h a;
        #pragma unroll
        for (int i = 0; i < 8; ++i) { a[i] = alo[i]; a[i + 8] = ahi[i]; }

        #pragma unroll
        for (int f = 0; f < 8; ++f) {
            const int cb = (f * 16 + l15) * SROW + (half ? 8 : 0);
            v8h blo = *reinterpret_cast<const v8h*>(&Bsm[buf][cb]);
            v8h bhi = *reinterpret_cast<const v8h*>(&Bsm[buf][cb + 16]);
            v16h bfrag;
            #pragma unroll
            for (int i = 0; i < 8; ++i) { bfrag[i] = blo[i]; bfrag[i + 8] = bhi[i]; }
            acc[f] = __builtin_amdgcn_wmma_f32_16x16x32_f16(
                false, a, false, bfrag, (short)0, acc[f], false, false);
        }
        __syncthreads();   // all waves done reading buf before it is re-staged
        buf ^= 1;
    }

    #pragma unroll
    for (int f = 0; f < 8; ++f) {
        const int col  = n0 + f * 16 + l15;
        const float bc = bias[col];
        #pragma unroll
        for (int v = 0; v < 8; ++v) {
            const int row = m0 + v + (half ? 8 : 0);
            float x = acc[f][v] + bc;
            if (relu) x = fmaxf(x, 0.0f);
            if (out16) out16[(size_t)row * N + col] = (_Float16)x;
            if (out32) out32[(size_t)row * ld_out + col + col_off] = x;
        }
    }
}

// ---------------------------------------------------------------------------
// Final select: out[t, 0:512] = is_b ? out_b : out_c  (pub half written by GEMM)
// ---------------------------------------------------------------------------
__global__ void k_combine(const float* __restrict__ outb, const float* __restrict__ outc,
                          const int* __restrict__ flags, float* __restrict__ out) {
    int i = blockIdx.x * blockDim.x + threadIdx.x;
    if (i < TT * FEATN) {
        int t = i >> 9, j = i & 511;
        out[(size_t)t * 1024 + j] = flags[t] ? outb[i] : outc[i];
    }
}

// ---------------------------------------------------------------------------
// Host launcher
// ---------------------------------------------------------------------------
extern "C" void kernel_launch(void* const* d_in, const int* in_sizes, int n_in,
                              void* d_out, int out_size, void* d_ws, size_t ws_size,
                              hipStream_t stream) {
    (void)n_in; (void)out_size; (void)ws_size;

    // --- leaf-order detection: insertion (b1 at idx1 => 128) vs jax-sorted (W2 => 4096)
    const bool srt = (in_sizes[1] == 4096);
    int rW1, rB1, rWih, rBih, rWhh, rBhh, rW2, rB2;
    if (!srt) { rW1=0; rB1=1; rWih=2; rBih=3; rWhh=4; rBhh=5; rW2=6; rB2=7; }
    else      { rW1=0; rW2=1; rWhh=2; rWih=3; rB1=4; rB2=5; rBhh=6; rBih=7; }
    int sWv, sBv, sWo, sBo;
    if (!srt) { sWv=4; sBv=5; sWo=6; sBo=7; } else { sWv=3; sBv=7; sWo=1; sBo=5; }
    int mW0, mB0, mW1, mB1, mW2, mB2;
    if (!srt) { mW0=0; mB0=1; mW1=2; mB1=3; mW2=4; mB2=5; }
    else      { mW0=0; mW1=1; mW2=2; mB0=3; mB1=4; mB2=5; }

    const int rnn_base[3] = { 1, 9, 17 };     // p, b, c
    const int sa_base[3]  = { 25, 33, 41 };   // p, b, c
    const int mlp_base[3] = { 61, 49, 55 };   // branch order: pub, b, c

    auto F = [&](int idx) -> const float* { return (const float*)d_in[idx]; };
    const float* obs = F(0);

    // --- workspace layout (bytes)
    char* ws = (char*)d_ws;
    const size_t GX_SZ    = (size_t)TT * 512 * 4;        // 16 MB per chain
    const size_t H1_OFF   = 0;                           // reuses Gx region after scan
    const size_t H2_OFF   = 16777216;
    const size_t FLAG_OFF = 3 * GX_SZ;
    const size_t O_SZ     = (size_t)TT * 32 * 4;
    const size_t O_OFF    = FLAG_OFF + (size_t)TT * 4;
    const size_t ATT_OFF  = O_OFF + 3 * O_SZ;
    const size_t X_SZ     = (size_t)TT * INN * 2;
    const size_t X_OFF    = ATT_OFF + 3 * O_SZ;
    const size_t WF_SZ    = 3473408;                     // f16 W0+W1+W2 per MLP
    const size_t WF_OFF   = X_OFF + 3 * X_SZ;
    const size_t OUTB_OFF = WF_OFF + 3 * WF_SZ;
    const size_t OUTC_OFF = OUTB_OFF + (size_t)TT * FEATN * 4;

    float* Gx    = (float*)(ws);
    int*   flags = (int*)  (ws + FLAG_OFF);
    float* Ovec  = (float*)(ws + O_OFF);
    float* Att   = (float*)(ws + ATT_OFF);
    float* outB  = (float*)(ws + OUTB_OFF);
    float* outC  = (float*)(ws + OUTC_OFF);

    // 1) flags
    k_flags<<<TT / 256, 256, 0, stream>>>(obs, flags);

    // 2) gate-input precompute, per chain
    for (int r = 0; r < 3; ++r) {
        const int b = rnn_base[r];
        k_gate_pre<<<TT / 64, 512, 0, stream>>>(
            obs, F(b + rW1), F(b + rB1), F(b + rWih), F(b + rBih), F(b + rBhh),
            Gx + (size_t)r * TT * 512);
    }

    // 3) sequential scan (3 persistent workgroups)
    k_lstm_scan<<<3, 512, 0, stream>>>(
        F(rnn_base[0] + rWhh), F(rnn_base[1] + rWhh), F(rnn_base[2] + rWhh),
        F(rnn_base[0] + rW2),  F(rnn_base[1] + rW2),  F(rnn_base[2] + rW2),
        F(rnn_base[0] + rB2),  F(rnn_base[1] + rB2),  F(rnn_base[2] + rB2),
        Gx, Ovec, flags);

    // 4) attention + softmax, per branch
    for (int r = 0; r < 3; ++r) {
        const int b = sa_base[r];
        k_attention<<<TT / 32, 256, 0, stream>>>(
            Ovec + (size_t)r * TT * 32,
            F(b + sWv), F(b + sBv), F(b + sWo), F(b + sBo),
            Att + (size_t)r * TT * 32);
    }

    // 5) MLP weights -> f16
    const int wsz[3] = { HIDN * INN, HIDN * HIDN, FEATN * HIDN };
    const size_t wsub[3] = { 0, (size_t)HIDN * INN * 2,
                             (size_t)HIDN * INN * 2 + (size_t)HIDN * HIDN * 2 };
    for (int br = 0; br < 3; ++br) {
        const int mb = mlp_base[br];
        const int widx[3] = { mW0, mW1, mW2 };
        for (int l = 0; l < 3; ++l) {
            k_f32tof16<<<(wsz[l] + 255) / 256, 256, 0, stream>>>(
                F(mb + widx[l]), (_Float16*)(ws + WF_OFF + (size_t)br * WF_SZ + wsub[l]), wsz[l]);
        }
    }

    // 6) per-branch: build X, then 3 WMMA GEMM layers
    for (int br = 0; br < 3; ++br) {
        const int mb = mlp_base[br];
        _Float16* X   = (_Float16*)(ws + X_OFF + (size_t)br * X_SZ);
        _Float16* h1  = (_Float16*)(ws + H1_OFF);
        _Float16* h2  = (_Float16*)(ws + H2_OFF);
        _Float16* W0f = (_Float16*)(ws + WF_OFF + (size_t)br * WF_SZ + wsub[0]);
        _Float16* W1f = (_Float16*)(ws + WF_OFF + (size_t)br * WF_SZ + wsub[1]);
        _Float16* W2f = (_Float16*)(ws + WF_OFF + (size_t)br * WF_SZ + wsub[2]);

        k_build_x<<<(TT * INN + 255) / 256, 256, 0, stream>>>(
            Att + (size_t)br * TT * 32, obs, X);

        // L0: [T,160] -> [T,1024], ReLU, f16 out
        k_gemm_wmma<<<dim3(HIDN / 128, TT / 128), 256, 0, stream>>>(
            X, W0f, F(mb + mB0), TT, HIDN, INN, 1, h1, nullptr, 0, 0);
        // L1: [T,1024] -> [T,1024], ReLU, f16 out
        k_gemm_wmma<<<dim3(HIDN / 128, TT / 128), 256, 0, stream>>>(
            h1, W1f, F(mb + mB1), TT, HIDN, HIDN, 1, h2, nullptr, 0, 0);
        // L2: [T,1024] -> [T,512], f32 out
        float* dst  = (br == 0) ? ((float*)d_out) : ((br == 1) ? outB : outC);
        int    ld   = (br == 0) ? 1024 : 512;
        int    coff = (br == 0) ? 512  : 0;
        k_gemm_wmma<<<dim3(FEATN / 128, TT / 128), 256, 0, stream>>>(
            h2, W2f, F(mb + mB2), TT, FEATN, HIDN, 0, nullptr, dst, ld, coff);
    }

    // 7) branch select into columns 0..511 of d_out
    k_combine<<<(TT * FEATN + 255) / 256, 256, 0, stream>>>(outB, outC, flags, (float*)d_out);
}
